// Attention_62130996904443
// MI455X (gfx1250) — matrix operand order repaired and checked by reading, not compiled, not run on hardware
//
#include <hip/hip_runtime.h>
#include <hip/hip_bf16.h>

#define DIM    1024
#define NHEADS 16
#define HDIM   64
#define BATCH  2
#define SEQ    2048
#define ROWS   (BATCH*SEQ)   // 4096 token rows

typedef __attribute__((ext_vector_type(16))) _Float16 v16h;
typedef __attribute__((ext_vector_type(8)))  _Float16 v8h;
typedef __attribute__((ext_vector_type(8)))  float    v8f;

// ---------------------------------------------------------------------------
// WMMA fragment loaders (layouts per CDNA5 ISA §7.12.2, wave32)
// A 16x32 f16:  lane m = lane&15, hi = lane>>4, half e -> K = (e/8)*16 + hi*8 + e%8
// B 32x16 f16:  lane n = lane&15, hi = lane>>4, half e -> K = (e/8)*16 + hi*8 + e%8
// C/D 16x16 f32: VGPR r -> M = r + 8*hi, N = lane&15
// ---------------------------------------------------------------------------
static __device__ inline v16h load_a_frag(const _Float16* __restrict__ A, int lda,
                                          int rowbase, int k0, int lane) {
  const int m = lane & 15, hi = lane >> 4;
  const _Float16* row = A + (size_t)(rowbase + m) * lda;
  v8h a0 = *(const v8h*)(row + k0 + hi * 8);        // K = k0 + hi*8 + 0..7
  v8h a1 = *(const v8h*)(row + k0 + 16 + hi * 8);   // K = k0 + 16 + hi*8 + 0..7
  v16h r;
#pragma unroll
  for (int i = 0; i < 8; ++i) { r[i] = a0[i]; r[i + 8] = a1[i]; }
  return r;
}

// B[k][n] = Mm[rowbase + n][k0 + k]  (second operand accessed K-contiguous:
// used for C = A*W^T, S = Q*K^T, and O = P*V with head-transposed V).
static __device__ inline v16h load_bT_frag(const _Float16* __restrict__ Mm, int ldm,
                                           int rowbase, int k0, int lane) {
  const int n = lane & 15, hi = lane >> 4;
  const _Float16* row = Mm + (size_t)(rowbase + n) * ldm + k0;
  v8h b0 = *(const v8h*)(row + hi * 8);
  v8h b1 = *(const v8h*)(row + 16 + hi * 8);
  v16h r;
#pragma unroll
  for (int i = 0; i < 8; ++i) { r[i] = b0[i]; r[i + 8] = b1[i]; }
  return r;
}

// ---------------------------------------------------------------------------
// f32 -> f16 conversion
// ---------------------------------------------------------------------------
__global__ void cvt_f32_f16(const float* __restrict__ src, _Float16* __restrict__ dst,
                            int nElem) {
  int i = blockIdx.x * blockDim.x + threadIdx.x;
  if (i < nElem) dst[i] = (_Float16)src[i];
}

// ---------------------------------------------------------------------------
// C(MxN) = A(MxK,f16) * W(NxK,f16)^T.  Wave tile: 64x32 (4x2 WMMA tiles).
// Block = 4 waves covering 64 rows x 128 cols; the 64 A-rows are SHARED by all
// 4 waves, so the A k-slab is staged into LDS (double-buffered) with
// global_load_async_to_lds_b128 (ASYNCcnt path, CDNA5 ISA).
// OUT_MODE: 0 = f16 row-major, 1 = f32 row-major, 2 = f16 head-transposed
//           (Vt[(b*1024 + col)][t], leading dim SEQ) for the V projection.
// ---------------------------------------------------------------------------
#define AS_STRIDE 56   // halves; 112 B row stride: 16B-aligned, conflict-free banks

template <int OUT_MODE>
__global__ __launch_bounds__(128) void wmma_gemm_xwt(
    const _Float16* __restrict__ A, const _Float16* __restrict__ W,
    _Float16* __restrict__ Ch, float* __restrict__ Cf, int N, int K) {
  __shared__ _Float16 As[2][64][AS_STRIDE];

  const int tid     = threadIdx.x;
  const int lane    = tid & 31;
  const int wave    = tid >> 5;
  const int rowbase = blockIdx.y * 64;
  const int colbase = blockIdx.x * 128 + wave * 32;

  // Cooperative async stage of A[rowbase..+63][k0..+31] into As[buf].
  // 128 threads x 2 chunks of 16B = 4 KB; per-lane LDS offset + global address.
  auto issue_async = [&](int buf, int k0) {
#pragma unroll
    for (int j = 0; j < 2; ++j) {
      const int c   = tid * 2 + j;          // 0..255
      const int row = c >> 2, ch = c & 3;   // 64 rows x 4 x 8-half chunks
      const unsigned long long g =
          (unsigned long long)(uintptr_t)(A + (size_t)(rowbase + row) * K + k0 + ch * 8);
      const unsigned l = (unsigned)(uintptr_t)(&As[buf][row][ch * 8]);
      asm volatile("global_load_async_to_lds_b128 %0, %1, off"
                   :: "v"(l), "v"(g) : "memory");
    }
  };

  v8f acc[4][2];
#pragma unroll
  for (int mi = 0; mi < 4; ++mi)
#pragma unroll
    for (int ni = 0; ni < 2; ++ni)
#pragma unroll
      for (int r = 0; r < 8; ++r) acc[mi][ni][r] = 0.0f;

  issue_async(0, 0);
  asm volatile("s_wait_asynccnt 0x0" ::: "memory");
  __syncthreads();

  const int m = lane & 15, hi = lane >> 4;
  for (int k0 = 0; k0 < K; k0 += 32) {
    const int cur = (k0 >> 5) & 1;
    if (k0 + 32 < K) issue_async(cur ^ 1, k0 + 32);

    // B fragments straight from global (distinct per wave)
    v16h b[2];
#pragma unroll
    for (int ni = 0; ni < 2; ++ni)
      b[ni] = load_bT_frag(W, K, colbase + ni * 16, k0, lane);

    // A fragments from LDS (ds_load_b128, shared across the 4 waves)
    v16h a[4];
#pragma unroll
    for (int mi = 0; mi < 4; ++mi) {
      const _Float16* base = &As[cur][mi * 16 + m][0];
      v8h a0 = *(const v8h*)(base + hi * 8);
      v8h a1 = *(const v8h*)(base + 16 + hi * 8);
#pragma unroll
      for (int i = 0; i < 8; ++i) { a[mi][i] = a0[i]; a[mi][i + 8] = a1[i]; }
    }

#pragma unroll
    for (int mi = 0; mi < 4; ++mi)
#pragma unroll
      for (int ni = 0; ni < 2; ++ni)
        acc[mi][ni] = __builtin_amdgcn_wmma_f32_16x16x32_f16(
            false, a[mi], false, b[ni], (short)0, acc[mi][ni], false, false);

    // my async writes into the next buffer are done + everyone finished
    // reading the current buffer -> safe to swap
    asm volatile("s_wait_asynccnt 0x0" ::: "memory");
    __syncthreads();
  }

  const int n = lane & 15;
#pragma unroll
  for (int mi = 0; mi < 4; ++mi)
#pragma unroll
    for (int ni = 0; ni < 2; ++ni)
#pragma unroll
      for (int r = 0; r < 8; ++r) {
        const int row = rowbase + mi * 16 + r + hi * 8;
        const int col = colbase + ni * 16 + n;
        const float v = acc[mi][ni][r];
        if (OUT_MODE == 0) {
          Ch[(size_t)row * N + col] = (_Float16)v;
        } else if (OUT_MODE == 1) {
          Cf[(size_t)row * N + col] = v;
        } else {                                   // head-transposed V store
          const int bb = row >> 11;                // row / SEQ
          const int t  = row & (SEQ - 1);
          Ch[((size_t)bb * (NHEADS * HDIM) + col) * SEQ + t] = (_Float16)v;
        }
      }
}

// ---------------------------------------------------------------------------
// Causal flash attention. One wave = one 16-row query tile of one (b,h).
// Q/K stored (ROWS x DIM) f16, head h in cols [h*64, h*64+64).
// V stored head-transposed: Vt[(b*1024 + h*64 + d)][t], leading dim SEQ.
// 64 keys per iteration; row sums via WMMA against a ones matrix (no shuffles);
// only the row max uses the 4-step 16-lane xor reduction.
// ---------------------------------------------------------------------------
__global__ __launch_bounds__(128) void wmma_flash_attn(
    const _Float16* __restrict__ Q, const _Float16* __restrict__ Km,
    const _Float16* __restrict__ Vt, _Float16* __restrict__ O) {
  __shared__ _Float16 Pl[4][16][72];          // per-wave 16x64 P tile, 144B row stride

  const int lane = threadIdx.x & 31;
  const int wave = threadIdx.x >> 5;
  const int tile = blockIdx.x * 4 + wave;     // 0 .. 4095
  const int qt   = tile & 127;                // query tile within (b,h); T/16 = 128
  const int h    = (tile >> 7) & 15;
  const int b    = tile >> 11;
  const int qbase  = qt * 16;
  const int rowoff = b * SEQ;
  const int hcol   = h * HDIM;
  const int vtrow  = b * (NHEADS * HDIM) + hcol;  // Vt row block for this (b,h)
  const int n = lane & 15, hi = lane >> 4;

  // Q fragments for the two K=32 chunks of head_dim=64
  v16h qa[2];
#pragma unroll
  for (int d = 0; d < 2; ++d)
    qa[d] = load_a_frag(Q, DIM, rowoff + qbase, hcol + d * 32, lane);

  v16h ones;
#pragma unroll
  for (int i = 0; i < 16; ++i) ones[i] = (_Float16)1.0f;

  v8f o[4];
  float mrow[8], lrow[8];
#pragma unroll
  for (int dt = 0; dt < 4; ++dt)
#pragma unroll
    for (int r = 0; r < 8; ++r) o[dt][r] = 0.0f;
#pragma unroll
  for (int r = 0; r < 8; ++r) { mrow[r] = -3.0e38f; lrow[r] = 0.0f; }

  const int kend = ((qbase + 16 + 63) / 64) * 64;   // causal: keys <= qbase+15 needed
  for (int kb = 0; kb < kend; kb += 64) {
    // ---- S(16x64) = Q * K^T ----
    v8f s[4];
#pragma unroll
    for (int nt = 0; nt < 4; ++nt) {
#pragma unroll
      for (int r = 0; r < 8; ++r) s[nt][r] = 0.0f;
#pragma unroll
      for (int d = 0; d < 2; ++d) {
        v16h bk = load_bT_frag(Km, DIM, rowoff + kb + nt * 16, hcol + d * 32, lane);
        s[nt] = __builtin_amdgcn_wmma_f32_16x16x32_f16(
            false, qa[d], false, bk, (short)0, s[nt], false, false);
      }
    }
    // ---- scale + causal mask (row = qbase + r + 8*hi, col = kb + nt*16 + n) ----
#pragma unroll
    for (int nt = 0; nt < 4; ++nt)
#pragma unroll
      for (int r = 0; r < 8; ++r) {
        const int qrow = qbase + r + hi * 8;
        const int kcol = kb + nt * 16 + n;
        s[nt][r] = (kcol > qrow) ? -3.0e38f : s[nt][r] * 0.125f;  // 1/sqrt(64)
      }
    // ---- row max: pairwise over 4 tiles, then 16-lane xor reduction ----
    float mnew[8], alpha[8];
#pragma unroll
    for (int r = 0; r < 8; ++r) {
      float v0 = fmaxf(fmaxf(s[0][r], s[1][r]), fmaxf(s[2][r], s[3][r]));
#pragma unroll
      for (int d = 1; d < 16; d <<= 1) v0 = fmaxf(v0, __shfl_xor(v0, d, 32));
      mnew[r]  = fmaxf(mrow[r], v0);
      alpha[r] = __expf(mrow[r] - mnew[r]);
      mrow[r]  = mnew[r];
    }
    // ---- P = exp(S - mnew) ----
#pragma unroll
    for (int nt = 0; nt < 4; ++nt)
#pragma unroll
      for (int r = 0; r < 8; ++r)
        s[nt][r] = __expf(s[nt][r] - mnew[r]);
    // ---- P: D layout -> LDS -> A layout (DS ops in-order within a wave) ----
#pragma unroll
    for (int nt = 0; nt < 4; ++nt)
#pragma unroll
      for (int r = 0; r < 8; ++r)
        Pl[wave][r + hi * 8][nt * 16 + n] = (_Float16)s[nt][r];

    v16h pa[2];
    {
      const _Float16* base = &Pl[wave][n][0];          // A-frag row m = lane&15
#pragma unroll
      for (int kk = 0; kk < 2; ++kk) {
        v8h p0 = *(const v8h*)(base + kk * 32 + hi * 8);
        v8h p1 = *(const v8h*)(base + kk * 32 + 16 + hi * 8);
#pragma unroll
        for (int i = 0; i < 8; ++i) { pa[kk][i] = p0[i]; pa[kk][i + 8] = p1[i]; }
      }
    }
    // ---- row sums via WMMA: P(16x64) * ones(64x16) -> every column = row sum ----
    v8f lsum;
#pragma unroll
    for (int r = 0; r < 8; ++r) lsum[r] = 0.0f;
#pragma unroll
    for (int kk = 0; kk < 2; ++kk)
      lsum = __builtin_amdgcn_wmma_f32_16x16x32_f16(
          false, pa[kk], false, ones, (short)0, lsum, false, false);
#pragma unroll
    for (int r = 0; r < 8; ++r) lrow[r] = lrow[r] * alpha[r] + lsum[r];
    // ---- rescale running output, then O += P * V ----
#pragma unroll
    for (int dt = 0; dt < 4; ++dt)
#pragma unroll
      for (int r = 0; r < 8; ++r) o[dt][r] *= alpha[r];
#pragma unroll
    for (int kk = 0; kk < 2; ++kk)
#pragma unroll
      for (int dt = 0; dt < 4; ++dt) {
        v16h vb = load_bT_frag(Vt, SEQ, vtrow + dt * 16, kb + kk * 32, lane);
        o[dt] = __builtin_amdgcn_wmma_f32_16x16x32_f16(
            false, pa[kk], false, vb, (short)0, o[dt], false, false);
      }
  }

  // ---- normalize rows and store f16 ----
#pragma unroll
  for (int r = 0; r < 8; ++r) lrow[r] = 1.0f / lrow[r];
#pragma unroll
  for (int dt = 0; dt < 4; ++dt)
#pragma unroll
    for (int r = 0; r < 8; ++r) {
      const int row = rowoff + qbase + r + hi * 8;
      const int col = hcol + dt * 16 + n;
      O[(size_t)row * DIM + col] = (_Float16)(o[dt][r] * lrow[r]);
    }
}

// ---------------------------------------------------------------------------
// Host launcher. Inputs: x, mask(unused; causal applied analytically),
// Wq, Wk, Wv, Wo. Output: (B,T,DIM) f32.
// ---------------------------------------------------------------------------
extern "C" void kernel_launch(void* const* d_in, const int* in_sizes, int n_in,
                              void* d_out, int out_size, void* d_ws, size_t ws_size,
                              hipStream_t stream) {
  (void)in_sizes; (void)n_in; (void)out_size; (void)ws_size;
  const float* x  = (const float*)d_in[0];
  const float* Wq = (const float*)d_in[2];
  const float* Wk = (const float*)d_in[3];
  const float* Wv = (const float*)d_in[4];
  const float* Wo = (const float*)d_in[5];
  float* out = (float*)d_out;

  char* ws = (char*)d_ws;
  size_t off = 0;
  _Float16* xh  = (_Float16*)(ws + off); off += (size_t)ROWS * DIM * 2;
  _Float16* Wqh = (_Float16*)(ws + off); off += (size_t)DIM * DIM * 2;
  _Float16* Wkh = (_Float16*)(ws + off); off += (size_t)DIM * DIM * 2;
  _Float16* Wvh = (_Float16*)(ws + off); off += (size_t)DIM * DIM * 2;
  _Float16* Woh = (_Float16*)(ws + off); off += (size_t)DIM * DIM * 2;
  _Float16* Qh  = (_Float16*)(ws + off); off += (size_t)ROWS * DIM * 2;
  _Float16* Kh  = (_Float16*)(ws + off); off += (size_t)ROWS * DIM * 2;
  _Float16* Vt  = (_Float16*)(ws + off); off += (size_t)ROWS * DIM * 2;  // head-transposed
  _Float16* Ah  = (_Float16*)(ws + off); off += (size_t)ROWS * DIM * 2;

  const int nx = ROWS * DIM, nw = DIM * DIM;
  cvt_f32_f16<<<(nx + 255) / 256, 256, 0, stream>>>(x,  xh,  nx);
  cvt_f32_f16<<<(nw + 255) / 256, 256, 0, stream>>>(Wq, Wqh, nw);
  cvt_f32_f16<<<(nw + 255) / 256, 256, 0, stream>>>(Wk, Wkh, nw);
  cvt_f32_f16<<<(nw + 255) / 256, 256, 0, stream>>>(Wv, Wvh, nw);
  cvt_f32_f16<<<(nw + 255) / 256, 256, 0, stream>>>(Wo, Woh, nw);

  dim3 gproj(DIM / 128, ROWS / 64);   // (8, 64)
  wmma_gemm_xwt<0><<<gproj, 128, 0, stream>>>(xh, Wqh, Qh, nullptr, DIM, DIM);
  wmma_gemm_xwt<0><<<gproj, 128, 0, stream>>>(xh, Wkh, Kh, nullptr, DIM, DIM);
  wmma_gemm_xwt<2><<<gproj, 128, 0, stream>>>(xh, Wvh, Vt, nullptr, DIM, DIM);

  // 4096 query tiles (B*H*T/16), 4 waves per block
  wmma_flash_attn<<<1024, 128, 0, stream>>>(Qh, Kh, Vt, Ah);

  wmma_gemm_xwt<1><<<gproj, 128, 0, stream>>>(Ah, Woh, nullptr, out, DIM, DIM);
}
